// HeteroDiffusionGraphForecaster_26834955665729
// MI455X (gfx1250) — matrix operand order repaired and checked by reading, not compiled, not run on hardware
//
#include <hip/hip_runtime.h>
#include <math.h>

// ---------------------------------------------------------------------------
// HeteroDiffusionGraphForecaster forward pass for MI455X (gfx1250, wave32).
//
// All K>=32 matmuls -> v_wmma_f32_16x16x32_bf16 (bf16 operands, fp32 accum).
// A tiles staged global->LDS with CDNA5 async-to-LDS DMA (ASYNCcnt) on the
// full-tile fast path; W tiles staged with vectorized loads + bf16 transpose.
// Tiny-K / tiny-O layers, 12x12 attention, LN/softmax stay scalar VALU.
// ---------------------------------------------------------------------------

#define Bc    16
#define Tc    12
#define Nc    400
#define TFc   3
#define AFc   4
#define EXOc  8
#define SFc   16
#define GHc   64
#define Dc    128
#define NHc   8
#define HDc   16
#define DFFc  512
#define HOUTc 12
#define BNc   (Bc*Nc)      // 6400
#define BTc   (Bc*Tc)      // 192
#define TBNc  (Bc*Tc*Nc)   // 76800

typedef __attribute__((ext_vector_type(16))) __bf16 v16bf;
typedef __attribute__((ext_vector_type(8)))  float  v8f;

__device__ __forceinline__ float actf(float x, int act) {
  switch (act) {
    case 1: return 0.5f * x * (1.0f + erff(x * 0.70710678118654752f)); // exact gelu
    case 2: return 1.0f / (1.0f + __expf(-x));                        // sigmoid
    case 3: return fmaxf(x, 0.0f);                                    // relu
    default: return x;
  }
}

// ---------------------------------------------------------------------------
// Generic batched WMMA GEMM:  C[g] = act( op(A[g]) @ W[g] + bias + resid )
//   op(A) = A (M x K row-major, lda) or A^T when TRANSA (A stored K-major).
//   W: K x N row-major (ldw).  C: M x N row-major (ldc).
// Block: 128 threads = 4 waves; block tile 64x64; wave tile 16x64; K-step 32.
// A tile kept in LDS as fp32 (async DMA friendly); converted to bf16 while
// building fragments. B tile converted to bf16 during the transpose-stage.
// ---------------------------------------------------------------------------
template <bool TRANSA>
__global__ __launch_bounds__(128)
void k_wmma_gemm(const float* __restrict__ A, long lda, long strA,
                 const float* __restrict__ Wm, long ldw, long strW,
                 const float* __restrict__ bias,
                 const float* __restrict__ resid,
                 float* __restrict__ C, long ldc, long strC,
                 int M, int N, int K, int accum, int act)
{
  __shared__ float  As[64][36];   // fp32 [m][k], row = 144B (16B multiple)
  __shared__ __bf16 Bs[64][40];   // bf16 [n][k] (W tile transposed)

  const int g = blockIdx.z;
  A  += (long)g * strA;
  Wm += (long)g * strW;
  C  += (long)g * strC;
  const float* R = resid ? (resid + (long)g * strC) : (const float*)0;

  const int m0 = blockIdx.x * 64;
  const int n0 = blockIdx.y * 64;
  const int tid  = threadIdx.x;
  const int lane = tid & 31;
  const int wv   = tid >> 5;
  const bool mfull = (m0 + 64 <= M);
  const bool nfull = (n0 + 64 <= N);

  v8f acc[4];
  #pragma unroll
  for (int nt = 0; nt < 4; ++nt)
    #pragma unroll
    for (int r = 0; r < 8; ++r) acc[nt][r] = 0.0f;

  for (int k0 = 0; k0 < K; k0 += 32) {
    const bool kfull = (k0 + 32 <= K);
    bool fastA = mfull && kfull;

    // ---------------- stage A tile (64 x 32 fp32) ----------------
    if (fastA) {
      if (TRANSA) {
        // A^T: read 4 consecutive m at fixed k, scatter into As rows.
        #pragma unroll
        for (int i = 0; i < 4; ++i) {
          int e  = tid + i * 128;       // chunk 0..511
          int k  = e >> 4;              // 0..31
          int mc = (e & 15) << 2;       // 0..60
          float4 v = *(const float4*)&A[(long)(k0 + k) * lda + (m0 + mc)];
          As[mc + 0][k] = v.x; As[mc + 1][k] = v.y;
          As[mc + 2][k] = v.z; As[mc + 3][k] = v.w;
        }
      } else {
        // Row-major A: k-contiguous 16B chunks -> async DMA global->LDS.
        #pragma unroll
        for (int i = 0; i < 4; ++i) {
          int e  = tid + i * 128;       // 16B chunk 0..511
          int m  = e >> 3;              // 0..63
          int kc = (e & 7) << 2;        // 0,4,..,28
          const float* gp = &A[(long)(m0 + m) * lda + (k0 + kc)];
          unsigned lo = (unsigned)(size_t)(&As[m][kc]);   // LDS byte offset
          asm volatile("global_load_async_to_lds_b128 %0, %1, off"
                       :: "v"(lo), "v"(gp) : "memory");
        }
      }
    } else {
      #pragma unroll
      for (int i = 0; i < 16; ++i) {
        int e = tid + i * 128;
        int m = TRANSA ? (e & 63) : (e >> 5);
        int k = TRANSA ? (e >> 6) : (e & 31);
        int gm = m0 + m, gk = k0 + k;
        float v = 0.0f;
        if (gm < M && gk < K)
          v = TRANSA ? A[(long)gk * lda + gm] : A[(long)gm * lda + gk];
        As[m][k] = v;
      }
    }

    // ---------------- stage W tile (32 x 64 -> Bs[n][k] bf16) ----------------
    if (nfull && kfull) {
      #pragma unroll
      for (int i = 0; i < 4; ++i) {
        int e  = tid + i * 128;         // chunk 0..511
        int k  = e >> 4;                // 0..31
        int nc = (e & 15) << 2;         // 0..60
        float4 w = *(const float4*)&Wm[(long)(k0 + k) * ldw + (n0 + nc)];
        Bs[nc + 0][k] = (__bf16)w.x; Bs[nc + 1][k] = (__bf16)w.y;
        Bs[nc + 2][k] = (__bf16)w.z; Bs[nc + 3][k] = (__bf16)w.w;
      }
      if ((k0 + 32) < K)  // prefetch next W k-tile (global_prefetch_b8)
        __builtin_prefetch(&Wm[(long)(k0 + 32 + (tid & 31)) * ldw + n0 + ((tid >> 5) << 4)], 0, 1);
    } else {
      #pragma unroll
      for (int i = 0; i < 16; ++i) {
        int e = tid + i * 128;
        int n = e & 63, k = e >> 6;
        int gn = n0 + n, gk = k0 + k;
        float v = 0.0f;
        if (gn < N && gk < K) v = Wm[(long)gk * ldw + gn];
        Bs[n][k] = (__bf16)v;
      }
    }

    if (fastA && !TRANSA)
      asm volatile("s_wait_asynccnt 0x0" ::: "memory");
    __syncthreads();

    // ---- A fragment: 16-bit A-matrix 16x32 layout, cvt fp32->bf16 here ----
    const int mr  = lane & 15;
    const int sel = (lane >> 4) << 3;
    const float* ar = &As[wv * 16 + mr][0];
    float4 a0 = *(const float4*)(ar + sel);
    float4 a1 = *(const float4*)(ar + sel + 4);
    float4 a2 = *(const float4*)(ar + 16 + sel);
    float4 a3 = *(const float4*)(ar + 20 + sel);
    union { v16bf v; __bf16 e[16]; } fa;
    fa.e[0]  = (__bf16)a0.x; fa.e[1]  = (__bf16)a0.y;
    fa.e[2]  = (__bf16)a0.z; fa.e[3]  = (__bf16)a0.w;
    fa.e[4]  = (__bf16)a1.x; fa.e[5]  = (__bf16)a1.y;
    fa.e[6]  = (__bf16)a1.z; fa.e[7]  = (__bf16)a1.w;
    fa.e[8]  = (__bf16)a2.x; fa.e[9]  = (__bf16)a2.y;
    fa.e[10] = (__bf16)a2.z; fa.e[11] = (__bf16)a2.w;
    fa.e[12] = (__bf16)a3.x; fa.e[13] = (__bf16)a3.y;
    fa.e[14] = (__bf16)a3.z; fa.e[15] = (__bf16)a3.w;

    #pragma unroll
    for (int nt = 0; nt < 4; ++nt) {
      // B fragment: 32x16 layout, lanes 0-15 K=0..7,16..23, lanes 16-31 K=8..15,24..31
      union { v16bf v; uint4 q[2]; } fb;
      const int nc = nt * 16 + mr;
      const int ko = (lane >> 4) << 3;
      fb.q[0] = *(const uint4*)&Bs[nc][ko];
      fb.q[1] = *(const uint4*)&Bs[nc][ko + 16];
      acc[nt] = __builtin_amdgcn_wmma_f32_16x16x32_bf16(
          false, fa.v, false, fb.v, (short)0, acc[nt], false, false);
    }
    __syncthreads();
  }

  // ---- epilogue: C/D 16x16 f32 layout (VGPR r -> M=r or 8+r) ----
  const int mrb = m0 + wv * 16 + ((lane >> 4) << 3);
  const int nc0 = n0 + (lane & 15);
  if (mfull && nfull) {
    #pragma unroll
    for (int nt = 0; nt < 4; ++nt) {
      int n = nc0 + nt * 16;
      float bv = bias ? bias[n] : 0.0f;
      #pragma unroll
      for (int r = 0; r < 8; ++r) {
        long ci = (long)(mrb + r) * ldc + n;
        float v = acc[nt][r] + bv;
        if (accum) v += C[ci];
        if (R) v += R[ci];
        C[ci] = actf(v, act);
      }
    }
  } else {
    #pragma unroll
    for (int nt = 0; nt < 4; ++nt) {
      int n = nc0 + nt * 16;
      if (n >= N) continue;
      #pragma unroll
      for (int r = 0; r < 8; ++r) {
        int m = mrb + r;
        if (m >= M) continue;
        float v = acc[nt][r];
        if (bias) v += bias[n];
        long ci = (long)m * ldc + n;
        if (accum) v += C[ci];
        if (R) v += R[ci];
        C[ci] = actf(v, act);
      }
    }
  }
}

// ---------------------------------------------------------------------------
// Scalar dense for tiny K / tiny O layers. rowmap==1 selects row (b, T-1, n).
// wt==1 means W is stored (O x K) row-major (used for src @ dst^T).
// ---------------------------------------------------------------------------
__global__ void k_dense_small(const float* __restrict__ in, long in_stride, int rowmap,
                              const float* __restrict__ Wm, int wt, int ldw,
                              const float* __restrict__ bias,
                              float* __restrict__ out, long out_stride,
                              int R, int K, int O, int act)
{
  long idx = (long)blockIdx.x * blockDim.x + threadIdx.x;
  if (idx >= (long)R * O) return;
  int r = (int)(idx / O), o = (int)(idx % O);
  long rin = r;
  if (rowmap == 1) {  // row (b, T-1, n) of a (B,T,N,*) tensor, r = b*N + n
    int b = r / Nc, n = r % Nc;
    rin = ((long)b * Tc + (Tc - 1)) * Nc + n;
  }
  const float* ip = in + rin * in_stride;
  float s = bias ? bias[o] : 0.0f;
  for (int k = 0; k < K; ++k)
    s += ip[k] * (wt ? Wm[(long)o * ldw + k] : Wm[(long)k * ldw + o]);
  out[(long)r * out_stride + o] = actf(s, act);
}

// ---------------------------------------------------------------------------
// LayerNorm: one wave32 per row (dim multiple of 32; here 64/128/256)
// ---------------------------------------------------------------------------
__global__ void k_layernorm(const float* __restrict__ in,
                            const float* __restrict__ gg,
                            const float* __restrict__ bb,
                            float* __restrict__ out, int rows, int dim)
{
  int row = blockIdx.x * 4 + (threadIdx.x >> 5);
  if (row >= rows) return;
  int lane = threadIdx.x & 31;
  const float* ip = in + (long)row * dim;
  int c = dim >> 5;
  float s = 0.0f, s2 = 0.0f;
  for (int i = 0; i < c; ++i) {
    float v = ip[lane + (i << 5)];
    s += v; s2 += v * v;
  }
  for (int o = 16; o > 0; o >>= 1) {
    s  += __shfl_xor(s,  o, 32);
    s2 += __shfl_xor(s2, o, 32);
  }
  float mean = s / dim;
  float inv  = rsqrtf(s2 / dim - mean * mean + 1e-5f);
  float* op = out + (long)row * dim;
  for (int i = 0; i < c; ++i) {
    int j = lane + (i << 5);
    op[j] = (ip[j] - mean) * inv * gg[j] + bb[j];
  }
}

// Row softmax in-place (wave32 per row), used for the 400x400 adaptive adj.
__global__ void k_softmax_rows(float* __restrict__ x, int rows, int cols)
{
  int row = blockIdx.x * 4 + (threadIdx.x >> 5);
  if (row >= rows) return;
  int lane = threadIdx.x & 31;
  float* p = x + (long)row * cols;
  float mx = -1e30f;
  for (int j = lane; j < cols; j += 32) mx = fmaxf(mx, p[j]);
  for (int o = 16; o > 0; o >>= 1) mx = fmaxf(mx, __shfl_xor(mx, o, 32));
  float s = 0.0f;
  for (int j = lane; j < cols; j += 32) { float e = __expf(p[j] - mx); p[j] = e; s += e; }
  for (int o = 16; o > 0; o >>= 1) s += __shfl_xor(s, o, 32);
  float inv = 1.0f / s;
  for (int j = lane; j < cols; j += 32) p[j] *= inv;
}

// ---------------------------------------------------------------------------
// MHA core over T=12 keys, hd=16: one thread per (bn, head, ti).
// Layout for q/k/v/out: row = t*BN + bn, col = h*hd + d.
// ---------------------------------------------------------------------------
__global__ void k_attn(const float* __restrict__ qp, const float* __restrict__ kp,
                       const float* __restrict__ vp, float* __restrict__ out)
{
  long idx = (long)blockIdx.x * blockDim.x + threadIdx.x;
  if (idx >= (long)BNc * NHc * Tc) return;
  int ti = (int)(idx % Tc);
  int t2 = (int)(idx / Tc);
  int hh = t2 % NHc;
  int bn = t2 / NHc;
  const float* qr = qp + ((long)ti * BNc + bn) * Dc + hh * HDc;
  float qv[HDc];
  #pragma unroll
  for (int d = 0; d < HDc; ++d) qv[d] = qr[d];
  float sc[Tc];
  float mx = -1e30f;
  #pragma unroll
  for (int tj = 0; tj < Tc; ++tj) {
    const float* kr = kp + ((long)tj * BNc + bn) * Dc + hh * HDc;
    float s = 0.0f;
    #pragma unroll
    for (int d = 0; d < HDc; ++d) s += qv[d] * kr[d];
    s *= 0.25f;  // 1/sqrt(16)
    sc[tj] = s;
    mx = fmaxf(mx, s);
  }
  float ssum = 0.0f;
  #pragma unroll
  for (int tj = 0; tj < Tc; ++tj) { sc[tj] = __expf(sc[tj] - mx); ssum += sc[tj]; }
  float inv = 1.0f / ssum;
  float av[HDc];
  #pragma unroll
  for (int d = 0; d < HDc; ++d) av[d] = 0.0f;
  #pragma unroll
  for (int tj = 0; tj < Tc; ++tj) {
    const float* vr = vp + ((long)tj * BNc + bn) * Dc + hh * HDc;
    float p = sc[tj] * inv;
    #pragma unroll
    for (int d = 0; d < HDc; ++d) av[d] += p * vr[d];
  }
  float* orow = out + ((long)ti * BNc + bn) * Dc + hh * HDc;
  #pragma unroll
  for (int d = 0; d < HDc; ++d) orow[d] = av[d];
}

// ---------------- elementwise glue ----------------
__global__ void k_mul(float* __restrict__ o, const float* __restrict__ a,
                      const float* __restrict__ b, long n)
{
  long i = (long)blockIdx.x * blockDim.x + threadIdx.x;
  if (i < n) o[i] = a[i] * b[i];
}

// ctx[btn, d] = base[bt, d] + accp[btn, d] + stp[n, d]
__global__ void k_ctx_combine(float* __restrict__ ctx, const float* __restrict__ base,
                              const float* __restrict__ accp, const float* __restrict__ stp)
{
  long i = (long)blockIdx.x * blockDim.x + threadIdx.x;
  if (i >= (long)TBNc * Dc) return;
  int d = (int)(i % Dc);
  long r = i / Dc;
  int bt = (int)(r / Nc);
  int n  = (int)(r % Nc);
  ctx[i] = base[(long)bt * Dc + d] + accp[i] + stp[(long)n * Dc + d];
}

// out[r,:] = x[r,:] * gate[r]
__global__ void k_rowgate(float* __restrict__ o, const float* __restrict__ x,
                          const float* __restrict__ gate, long rows, int dim)
{
  long i = (long)blockIdx.x * blockDim.x + threadIdx.x;
  if (i >= rows * dim) return;
  o[i] = x[i] * gate[i / dim];
}

// mix = (init ? 0 : mix) + gates[r,gi] * term
__global__ void k_mix(float* __restrict__ mix, const float* __restrict__ term,
                      const float* __restrict__ gates, int gi, int init)
{
  long i = (long)blockIdx.x * blockDim.x + threadIdx.x;
  if (i >= (long)TBNc * GHc) return;
  long r = i / GHc;
  float g = gates[r * 4 + gi];
  mix[i] = (init ? 0.0f : mix[i]) + g * term[i];
}

// tmp = h + gelu(mix * scale)
__global__ void k_diff_post(float* __restrict__ tmp, const float* __restrict__ h,
                            const float* __restrict__ mix, const float* __restrict__ sc)
{
  long i = (long)blockIdx.x * blockDim.x + threadIdx.x;
  if (i >= (long)TBNc * GHc) return;
  float m = mix[i] * sc[i];
  tmp[i] = h[i] + 0.5f * m * (1.0f + erff(m * 0.70710678118654752f));
}

// q[t,bn,:] = hm[b,t,n,:] + pe(t,:) ; kv[t,bn,:] = ctx[b,t,n,:]
__global__ void k_build_qkv(const float* __restrict__ hm, const float* __restrict__ ctx,
                            float* __restrict__ q, float* __restrict__ kv)
{
  long i = (long)blockIdx.x * blockDim.x + threadIdx.x;
  if (i >= (long)Tc * BNc * Dc) return;
  int d = (int)(i % Dc);
  long r = i / Dc;
  int bn = (int)(r % BNc);
  int t  = (int)(r / BNc);
  int b = bn / Nc, n = bn % Nc;
  long src = (((long)b * Tc + t) * Nc + n) * Dc + d;
  int hi = d >> 1;
  float div = __expf(-9.210340371976184f * (float)(2 * hi) / (float)Dc);
  float ang = (float)t * div;
  float pe = (d & 1) ? __cosf(ang) : __sinf(ang);
  q[i]  = hm[src] + pe;
  kv[i] = ctx[src];
}

// feat[bn, 0:128] = zlast[bn,:], feat[bn, 128:256] = latest[bn,:]
__global__ void k_concat(float* __restrict__ feat, const float* __restrict__ zlast,
                         const float* __restrict__ latest)
{
  long i = (long)blockIdx.x * blockDim.x + threadIdx.x;
  if (i >= (long)BNc * 2 * Dc) return;
  int d = (int)(i % (2 * Dc));
  long bn = i / (2 * Dc);
  feat[i] = (d < Dc) ? zlast[bn * Dc + d] : latest[bn * Dc + (d - Dc)];
}

// ---------------------------------------------------------------------------
// Parameter leaf indices (alphabetical dict flattening, lists in order):
//  0 x_traffic, 1 a_norm, 2 x_exo, 3 x_static
//  4 acc1.W 5 acc1.b 6 acc2.W 7 acc2.b
//  8..97  blocks[0..2], 30 leaves each:
//    +0 adp0.W +1 adp0.b +2 adp1.W +3 adp1.b
//    +4 bwd0.W +5 bwd0.b +6 bwd1.W +7 bwd1.b
//    +8 cg1.W +9 cg1.b +10 cg2.W +11 cg2.b
//    +12 cs1.W +13 cs1.b +14 cs2.W +15 cs2.b
//    +16 dg.W +17 dg.b +18 dst
//    +19 fwd0.W +20 fwd0.b +21 fwd1.W +22 fwd1.b
//    +23 ln.b +24 ln.g +25 self.W +26 self.b +27 sg.W +28 sg.b +29 src
//  98 cross.Wo 99 cross.Wqkv 100 cross.bo 101 cross.bqkv
//  102 cross_ln.b 103 cross_ln.g 104 ctx2hid.W 105 ctx2hid.b
//  106 exo1.W 107 exo1.b 108 exo2.W 109 exo2.b 110 exo_gate.W 111 exo_gate.b
//  112 latest.W 113 latest.b 114 out1.W 115 out1.b 116 out2.W 117 out2.b
//  118 out_ln.b 119 out_ln.g 120 st1.W 121 st1.b 122 st2.W 123 st2.b
//  124..147 tf[0..1], 12 leaves each:
//    +0 attn.Wo +1 attn.Wqkv +2 attn.bo +3 attn.bqkv
//    +4 ff1.W +5 ff1.b +6 ff2.W +7 ff2.b +8 ln1.b +9 ln1.g +10 ln2.b +11 ln2.g
//  148 traffic_proj.W 149 traffic_proj.b 150 traffic_to_model.W 151 .b
// ---------------------------------------------------------------------------

extern "C" void kernel_launch(void* const* d_in, const int* in_sizes, int n_in,
                              void* d_out, int out_size, void* d_ws, size_t ws_size,
                              hipStream_t stream)
{
  (void)in_sizes; (void)n_in; (void)out_size; (void)ws_size;
  auto F = [&](int i) -> const float* { return (const float*)d_in[i]; };
  const float* xtr  = F(0);
  const float* an   = F(1);
  const float* xexo = F(2);
  const float* xst  = F(3);

  // ---------------- workspace arena (floats) ----------------
  float* ws = (float*)d_ws;
  const size_t SL = (size_t)TBNc * Dc;  // 9,830,400 floats per big slot
  float* ctx = ws;                      // context (B,T,N,D)
  float* hb  = ctx + SL;                // h (B,T,N,GH)
  float* qb  = hb + (size_t)TBNc * GHc; // q (T,BN,D)
  float* kvb = qb + SL;                 // kv (T,BN,D)
  float* sA  = kvb + SL;                // scratch slots (each SL floats)
  float* sB  = sA + SL;
  float* sC  = sB + SL;
  float* sD  = sC + SL;
  float* sE  = sD + SL;

  // diffusion-stage aliases inside scratch slots
  float* d_tmp   = sA;
  float* d_scale = sA + (size_t)TBNc * GHc;
  float* d_term  = sB;
  float* d_mix   = sB + (size_t)TBNc * GHc;
  float* d_s0    = sC;
  float* d_s1    = sC + (size_t)TBNc * GHc;
  float* gates   = sD;
  float* sgv     = sD + (size_t)TBNc * 4;
  float* dgv     = sgv + TBNc;
  float* adpS    = dgv + TBNc;          // 400x400

  // context-build aliases (sE)
  float* e_tmp   = sE;
  float* e_base  = sE + (size_t)BTc * Dc;
  float* e_gate  = sE + 2 * (size_t)BTc * Dc;
  float* ctxbase = sE + 3 * (size_t)BTc * Dc;
  float* st_t    = sE + 4 * (size_t)BTc * Dc;
  float* st_o    = st_t + (size_t)Nc * Dc;

  // ---------------- launch helpers ----------------
  auto gemm = [&](const float* A, long lda, long strA, int tA,
                  const float* Wm, long ldw, long strW,
                  const float* bias, const float* resid,
                  float* C, long ldc, long strC,
                  int M, int N, int K, int batches, int accum, int act) {
    dim3 g((M + 63) / 64, (N + 63) / 64, batches);
    if (tA)
      k_wmma_gemm<true><<<g, dim3(128), 0, stream>>>(A, lda, strA, Wm, ldw, strW,
                                                     bias, resid, C, ldc, strC,
                                                     M, N, K, accum, act);
    else
      k_wmma_gemm<false><<<g, dim3(128), 0, stream>>>(A, lda, strA, Wm, ldw, strW,
                                                      bias, resid, C, ldc, strC,
                                                      M, N, K, accum, act);
  };
  auto dsmall = [&](const float* in, long instr, int rowmap,
                    const float* Wm, int wt, int ldw, const float* bias,
                    float* out, long outstr, int R, int K, int O, int act) {
    long tot = (long)R * O;
    k_dense_small<<<(int)((tot + 255) / 256), 256, 0, stream>>>(
        in, instr, rowmap, Wm, wt, ldw, bias, out, outstr, R, K, O, act);
  };
  auto lnorm = [&](const float* in, const float* g, const float* b,
                   float* out, int rows, int dim) {
    k_layernorm<<<(rows + 3) / 4, 128, 0, stream>>>(in, g, b, out, rows, dim);
  };
  auto nb = [](long n) { return (int)((n + 255) / 256); };

  // ================= context =================
  // exo path (192 rows)
  dsmall(xexo, EXOc, 0, F(106), 0, Dc, F(107), e_tmp, Dc, BTc, EXOc, Dc, 1);  // exo1+gelu
  dsmall(e_tmp, Dc, 0, F(108), 0, Dc, F(109), e_base, Dc, BTc, Dc, Dc, 0);    // exo2
  dsmall(xexo, EXOc, 0, F(110), 0, Dc, F(111), e_gate, Dc, BTc, EXOc, Dc, 2); // exo_gate sigmoid
  k_mul<<<nb((long)BTc * Dc), 256, 0, stream>>>(ctxbase, e_base, e_gate, (long)BTc * Dc);

  // acc path (76800 rows): acc1 (K=4) scalar, acc2 (128x128) WMMA
  dsmall(xtr + TFc, TFc + AFc, 0, F(4), 0, Dc, F(5), sB, Dc, TBNc, AFc, Dc, 1);
  gemm(sB, Dc, 0, 0, F(6), Dc, 0, F(7), 0, sA, Dc, 0, TBNc, Dc, Dc, 1, 0, 0);

  // static path (400 rows)
  dsmall(xst, SFc, 0, F(120), 0, Dc, F(121), st_t, Dc, Nc, SFc, Dc, 1);
  gemm(st_t, Dc, 0, 0, F(122), Dc, 0, F(123), 0, st_o, Dc, 0, Nc, Dc, Dc, 1, 0, 0);

  k_ctx_combine<<<nb((long)TBNc * Dc), 256, 0, stream>>>(ctx, ctxbase, sA, st_o);

  // ================= h init =================
  dsmall(xtr, TFc + AFc, 0, F(148), 0, GHc, F(149), hb, GHc, TBNc, TFc, GHc, 0); // traffic_proj
  gemm(ctx, Dc, 0, 0, F(104), GHc, 0, F(105), 0, hb, GHc, 0, TBNc, GHc, Dc, 1, 1, 0); // += ctx2hid

  // ================= diffusion blocks =================
  for (int blk = 0; blk < 3; ++blk) {
    int bp = 8 + blk * 30;
    // gates / scale / sg / dg from context
    gemm(ctx, Dc, 0, 0, F(bp + 8), GHc, 0, F(bp + 9), 0, d_tmp, GHc, 0, TBNc, GHc, Dc, 1, 0, 1);  // cg1+gelu
    dsmall(d_tmp, GHc, 0, F(bp + 10), 0, 4, F(bp + 11), gates, 4, TBNc, GHc, 4, 2);               // cg2 sigmoid
    gemm(ctx, Dc, 0, 0, F(bp + 12), GHc, 0, F(bp + 13), 0, d_tmp, GHc, 0, TBNc, GHc, Dc, 1, 0, 1); // cs1+gelu
    gemm(d_tmp, GHc, 0, 0, F(bp + 14), GHc, 0, F(bp + 15), 0, d_scale, GHc, 0, TBNc, GHc, GHc, 1, 0, 2); // cs2 sigmoid
    dsmall(ctx, Dc, 0, F(bp + 27), 0, 1, F(bp + 28), sgv, 1, TBNc, Dc, 1, 2);                     // sg
    dsmall(ctx, Dc, 0, F(bp + 16), 0, 1, F(bp + 17), dgv, 1, TBNc, Dc, 1, 2);                     // dg
    // adaptive adjacency: relu(src @ dst^T) then row-softmax
    dsmall(F(bp + 29), 16, 0, F(bp + 18), 1, 16, 0, adpS, Nc, Nc, 16, Nc, 3);
    k_softmax_rows<<<(Nc + 3) / 4, 128, 0, stream>>>(adpS, Nc, Nc);

    // self term
    gemm(hb, GHc, 0, 0, F(bp + 25), GHc, 0, F(bp + 26), 0, d_term, GHc, 0, TBNc, GHc, GHc, 1, 0, 0);
    k_mix<<<nb((long)TBNc * GHc), 256, 0, stream>>>(d_mix, d_term, gates, 0, 1);

    const long sBT = (long)Nc * GHc;  // per-(b,t) batch stride
    // fwd branch: state = h*dg, two hops of adj
    k_rowgate<<<nb((long)TBNc * GHc), 256, 0, stream>>>(d_s0, hb, dgv, TBNc, GHc);
    gemm(an, Nc, 0, 0, d_s0, GHc, sBT, 0, 0, d_s1, GHc, sBT, Nc, GHc, Nc, BTc, 0, 0);
    gemm(d_s1, GHc, 0, 0, F(bp + 19), GHc, 0, F(bp + 20), 0, d_term, GHc, 0, TBNc, GHc, GHc, 1, 0, 0);
    gemm(an, Nc, 0, 0, d_s1, GHc, sBT, 0, 0, d_s0, GHc, sBT, Nc, GHc, Nc, BTc, 0, 0);
    gemm(d_s0, GHc, 0, 0, F(bp + 21), GHc, 0, F(bp + 22), 0, d_term, GHc, 0, TBNc, GHc, GHc, 1, 1, 0);
    k_mix<<<nb((long)TBNc * GHc), 256, 0, stream>>>(d_mix, d_term, gates, 1, 0);

    // bwd branch: state = h*sg, adj transposed
    k_rowgate<<<nb((long)TBNc * GHc), 256, 0, stream>>>(d_s0, hb, sgv, TBNc, GHc);
    gemm(an, Nc, 0, 1, d_s0, GHc, sBT, 0, 0, d_s1, GHc, sBT, Nc, GHc, Nc, BTc, 0, 0);
    gemm(d_s1, GHc, 0, 0, F(bp + 4), GHc, 0, F(bp + 5), 0, d_term, GHc, 0, TBNc, GHc, GHc, 1, 0, 0);
    gemm(an, Nc, 0, 1, d_s1, GHc, sBT, 0, 0, d_s0, GHc, sBT, Nc, GHc, Nc, BTc, 0, 0);
    gemm(d_s0, GHc, 0, 0, F(bp + 6), GHc, 0, F(bp + 7), 0, d_term, GHc, 0, TBNc, GHc, GHc, 1, 1, 0);
    k_mix<<<nb((long)TBNc * GHc), 256, 0, stream>>>(d_mix, d_term, gates, 2, 0);

    // adp branch: state = h, learned adjacency
    gemm(adpS, Nc, 0, 0, hb, GHc, sBT, 0, 0, d_s1, GHc, sBT, Nc, GHc, Nc, BTc, 0, 0);
    gemm(d_s1, GHc, 0, 0, F(bp + 0), GHc, 0, F(bp + 1), 0, d_term, GHc, 0, TBNc, GHc, GHc, 1, 0, 0);
    gemm(adpS, Nc, 0, 0, d_s1, GHc, sBT, 0, 0, d_s0, GHc, sBT, Nc, GHc, Nc, BTc, 0, 0);
    gemm(d_s0, GHc, 0, 0, F(bp + 2), GHc, 0, F(bp + 3), 0, d_term, GHc, 0, TBNc, GHc, GHc, 1, 1, 0);
    k_mix<<<nb((long)TBNc * GHc), 256, 0, stream>>>(d_mix, d_term, gates, 3, 0);

    // h = LN(h + gelu(mix * scale))
    k_diff_post<<<nb((long)TBNc * GHc), 256, 0, stream>>>(d_tmp, hb, d_mix, d_scale);
    lnorm(d_tmp, F(bp + 24), F(bp + 23), hb, TBNc, GHc);
  }

  // ================= to model dim + q/kv =================
  gemm(hb, GHc, 0, 0, F(150), Dc, 0, F(151), 0, sA, Dc, 0, TBNc, Dc, GHc, 1, 0, 0); // hm -> sA
  k_build_qkv<<<nb((long)Tc * BNc * Dc), 256, 0, stream>>>(sA, ctx, qb, kvb);

  // ================= cross attention =================
  {
    const float* Wqkv = F(99);
    const float* bqkv = F(101);
    gemm(qb,  Dc, 0, 0, Wqkv,        384, 0, bqkv,        0, sA, Dc, 0, TBNc, Dc, Dc, 1, 0, 0); // Q
    gemm(kvb, Dc, 0, 0, Wqkv + 128,  384, 0, bqkv + 128,  0, sB, Dc, 0, TBNc, Dc, Dc, 1, 0, 0); // K
    gemm(kvb, Dc, 0, 0, Wqkv + 256,  384, 0, bqkv + 256,  0, sC, Dc, 0, TBNc, Dc, Dc, 1, 0, 0); // V
    k_attn<<<nb((long)BNc * NHc * Tc), 256, 0, stream>>>(sA, sB, sC, sD);
    gemm(sD, Dc, 0, 0, F(98), Dc, 0, F(100), qb, sE, Dc, 0, TBNc, Dc, Dc, 1, 0, 0); // Wo + resid(q)
    lnorm(sE, F(103), F(102), qb, TBNc, Dc);
  }

  // ================= transformer layers =================
  for (int l = 0; l < 2; ++l) {
    int tb = 124 + l * 12;
    const float* Wqkv = F(tb + 1);
    const float* bqkv = F(tb + 3);
    lnorm(qb, F(tb + 9), F(tb + 8), sE, TBNc, Dc);  // ln1
    gemm(sE, Dc, 0, 0, Wqkv,       384, 0, bqkv,       0, sA, Dc, 0, TBNc, Dc, Dc, 1, 0, 0);
    gemm(sE, Dc, 0, 0, Wqkv + 128, 384, 0, bqkv + 128, 0, sB, Dc, 0, TBNc, Dc, Dc, 1, 0, 0);
    gemm(sE, Dc, 0, 0, Wqkv + 256, 384, 0, bqkv + 256, 0, sC, Dc, 0, TBNc, Dc, Dc, 1, 0, 0);
    k_attn<<<nb((long)BNc * NHc * Tc), 256, 0, stream>>>(sA, sB, sC, sD);
    gemm(sD, Dc, 0, 0, F(tb + 0), Dc, 0, F(tb + 2), qb, qb, Dc, 0, TBNc, Dc, Dc, 1, 0, 0); // q += attn proj
    lnorm(qb, F(tb + 11), F(tb + 10), sE, TBNc, Dc);  // ln2
    // FF: fbuf (76800 x 512) spans sA..sD
    gemm(sE, Dc, 0, 0, F(tb + 4), DFFc, 0, F(tb + 5), 0, sA, DFFc, 0, TBNc, DFFc, Dc, 1, 0, 1);   // ff1+gelu
    gemm(sA, DFFc, 0, 0, F(tb + 6), Dc, 0, F(tb + 7), qb, qb, Dc, 0, TBNc, Dc, DFFc, 1, 0, 0);    // q += ff2
  }

  // ================= head =================
  const float* zlast = qb + (long)(Tc - 1) * BNc * Dc;
  dsmall(xtr, TFc + AFc, 1, F(112), 0, Dc, F(113), sB, Dc, BNc, TFc, Dc, 0);  // latest
  k_concat<<<nb((long)BNc * 2 * Dc), 256, 0, stream>>>(sC, zlast, sB);         // feat (6400 x 256)
  lnorm(sC, F(119), F(118), sD, BNc, 2 * Dc);                                  // out_ln
  gemm(sD, 2 * Dc, 0, 0, F(114), Dc, 0, F(115), 0, sE, Dc, 0, BNc, Dc, 2 * Dc, 1, 0, 1); // out1+gelu
  dsmall(sE, Dc, 0, F(116), 0, HOUTc, F(117), (float*)d_out, HOUTc, BNc, Dc, HOUTc, 0);  // out2 -> d_out
}
